// PinkNet_11897059409922
// MI455X (gfx1250) — matrix operand order repaired and checked by reading, not compiled, not run on hardware
//
#include <hip/hip_runtime.h>
#include <stdint.h>

// ---------------------------------------------------------------------------
// PinkNet on MI455X (gfx1250), exploiting the exact FixedLinear mask structure
//   mask[o,t,y,x] = (x even) && (y%2 == (o+1)%2)    (leftover==0, no RNG used)
// Both convs as implicit GEMM on v_wmma_f32_16x16x32_f16.
//
// Workspace:
//  P1 : conv1 input, x-parity phase px=0 only (only odd input xx is ever read
//       by kept outputs), zero-padded f16, octet packed:
//       [t_p(19)][py(2)][v_in(31)][uo(4)][b(512)][8]
//  H0 : kept conv1 outputs (compressed 4x), f16, octet packed:
//       [c(4)][t_p(19)][ykc(22)][wo(3)][b(512)][8]   (w_p = u+5)
//  H1 : same data shifted by 6 w-slots (w_p2 = u-1, u>=1) so conv2's second
//       x-tile window is octet aligned: [c][t_p][ykc(22)][wo(2)][b][8]
//  A1 : 400 conv1 A-frags (kt,ky,py), banded weights, contiguous k = 16*(l>>4) + s
//  A2 : 400 conv2 A-frags (c,kt,seg,q): K=32 = two kept-ky halves x 16 w-cols,
//       A[m][k] = w2[c,kt, q+4s+2*(k>>4), 2*(k&15) - m]  (same for both tiles)
//
// WMMA register layout assumptions (CDNA5 ISA 7.12.2):
//  A 16x32 f16 : lane l -> M = l&15 ; K-pair(lane,dword d) =
//                d + ((l>>4)<<3)   (contiguous k = 16*(l>>4) + s, same as B)
//  B 32x16 f16 : lane l -> N = l&15 ; lanes 0-15 hold K=0..15,
//                lanes 16-31 hold K=16..31, 2 packed K per dword (low=even K); this code's order, not the ISA table
//  C/D 16x16 f32: lane l, VGPR r -> M = r + 8*(l>>4), N = l&15
// ---------------------------------------------------------------------------

typedef _Float16 v16h __attribute__((ext_vector_type(16)));
typedef float    v8f  __attribute__((ext_vector_type(8)));

union Frag16 {
    v16h v;
    uint4 q4[2];
    uint32_t d[8];
};

__device__ __forceinline__ uint16_t f32_to_f16u(float f) {
    union { _Float16 h; uint16_t u; } c;
    c.h = (_Float16)f;           // v_cvt_f16_f32 (RNE)
    return c.u;
}

#define WMMA16(af, bf, acc) \
    __builtin_amdgcn_wmma_f32_16x16x32_f16(false, (af).v, false, (bf).v, \
                                           (short)0, (acc), false, false)

// ---- workspace layout (bytes) ----
#define P1_BYTES   (19u*2u*31u*4u*4096u*2u)   /* 38,600,704  */
#define SLACK      65536u
#define H0_OFF     (P1_BYTES + SLACK)         /* 38,666,240  */
#define H0_BYTES   (4u*19u*22u*3u*4096u*2u)   /* 41,091,072  */
#define H1_OFF     (H0_OFF + H0_BYTES)        /* 79,757,312  */
#define H1_BYTES   (4u*19u*22u*2u*4096u*2u)   /* 27,394,048  */
#define A1_OFF     (H1_OFF + H1_BYTES)        /* 107,151,360 */
#define A1_BYTES   (400u*256u*4u)
#define A2_OFF     (A1_OFF + A1_BYTES)
#define A2_BYTES   (400u*256u*4u)
#define MEMSET_BYTES A1_OFF                    /* zero P1+slack+H0+H1 */

// ---------------------------------------------------------------------------
// Pack fp32 input (B,X,Y,T,1) -> padded phase layout P1 (odd xx columns only).
// grid: 10*24*12*512/256 = 5760 blocks.
// ---------------------------------------------------------------------------
__global__ void prep_p1(const float* __restrict__ x, uint16_t* __restrict__ P1) {
    int j  = blockIdx.x * 256 + threadIdx.x;      // < 1,474,560
    int b  = j & 511;
    int r  = j >> 9;                               // ((t*24+yy)*12+xi)
    int xi = r % 12;
    int yy = (r / 12) % 24;
    int t  = r / 288;
    int xx = 2 * xi + 1;                           // odd input columns only
    float f = x[((b * 24 + xx) * 24 + yy) * 10 + t];
    int ypf = yy + 19, xpf = xx + 19;              // xpf even (px=0 phase)
    int py = ypf & 1;
    int v  = ypf >> 1, u = xpf >> 1;               // u in [10,21]
    int e = (((((t + 7) * 2 + py) * 31 + v) * 4 + (u >> 3))) * 4096 + b * 8 + (u & 7);
    P1[e] = f32_to_f16u(f);
}

// ---------------------------------------------------------------------------
// Conv1 A-frags: 400 ((kt*20+ky)*2+py). Rows m: oi=m>>3 -> o=2*oi+(1-py),
// du=m&7. A[m][k] = w1[o,kt,ky, k-du]. Shared by both u-tiles (window shifts).
// ---------------------------------------------------------------------------
__global__ void prep_a1(const float* __restrict__ w1, uint32_t* __restrict__ A1) {
    int f  = blockIdx.x;
    int py = f & 1;
    int ky = (f >> 1) % 20;
    int kt = f / 40;
    int tid = threadIdx.x;
    int l = tid >> 3, d = tid & 7;
    int m = l & 15;
    int oi = m >> 3, du = m & 7;
    int o  = 2 * oi + (1 - py);
    int kp = d + ((l >> 4) << 3);
    uint32_t outv = 0;
    for (int e = 0; e < 2; ++e) {
        int k  = 2 * kp + e;
        int kx = k - du;
        float w = (kx >= 0 && kx < 20) ? w1[o * 4000 + kt * 400 + ky * 20 + kx] : 0.0f;
        outv |= (uint32_t)f32_to_f16u(w) << (16 * e);
    }
    A1[f * 256 + tid] = outv;
}

// ---------------------------------------------------------------------------
// Conv2 A-frags: 400 (((c*10+kt)*5+s)*2+q). Rows m = x offset (12 used).
// k-half selects ky = q+4s+2*(k>>4); kx = 2*(k&15) - m (identical both tiles).
// ---------------------------------------------------------------------------
__global__ void prep_a2(const float* __restrict__ w2, uint32_t* __restrict__ A2) {
    int f = blockIdx.x;
    int q = f & 1;
    int rest = f >> 1;
    int s  = rest % 5;
    int ck = rest / 5;
    int kt = ck % 10;
    int c  = ck / 10;
    int tid = threadIdx.x;
    int l = tid >> 3, d = tid & 7;
    int m = l & 15;
    int kp = d + ((l >> 4) << 3);
    uint32_t outv = 0;
    for (int e = 0; e < 2; ++e) {
        int k  = 2 * kp + e;
        int khalf = k >> 4, kw = k & 15;
        int ky = q + 4 * s + 2 * khalf;
        int kx = 2 * kw - m;
        float w = (m < 12 && kx >= 0 && kx < 20)
                    ? w2[c * 4000 + kt * 400 + ky * 20 + kx] : 0.0f;
        outv |= (uint32_t)f32_to_f16u(w) << (16 * e);
    }
    A2[f * 256 + tid] = outv;
}

// ---------------------------------------------------------------------------
// Conv1 implicit GEMM over kept outputs only. Rows = 2 kept ch x 8 phase-x;
// two u-tiles (s=0,s=8) share one A frag and overlap one octet: per (kt,ky)
// 6 B-b128 + 2 A-b128 -> 4 WMMAs. Fused ReLU+mask, store compressed H0+H1.
// grid = 240 (t,py,v) groups * 2 batch-groups; 8 waves * 32 batch.
// ---------------------------------------------------------------------------
__global__ void __launch_bounds__(256)
conv1_gemm(const uint16_t* __restrict__ P1,
           const uint32_t* __restrict__ A1,
           const float* __restrict__ mask,
           uint16_t* __restrict__ H0, uint16_t* __restrict__ H1) {
    int rg = blockIdx.x >> 1, bg = blockIdx.x & 1;
    int v  = rg % 12;
    int py = (rg / 12) & 1;
    int t  = rg / 24;

    int l  = threadIdx.x & 31;
    int wv = threadIdx.x >> 5;
    int b0 = bg * 256 + wv * 32;
    int n  = l & 15, h = l >> 4;
    int lo = h * 8192 + (b0 + n) * 8;   // lane element offset in a P1 row

    v8f acc[2][2];
#pragma unroll
    for (int i = 0; i < 2; ++i)
#pragma unroll
        for (int j2 = 0; j2 < 2; ++j2)
            acc[i][j2] = (v8f){0.f,0.f,0.f,0.f,0.f,0.f,0.f,0.f};

    const uint32_t* Al = A1 + l * 8;

    for (int kt = 0; kt < 10; ++kt) {
        int tp = t + kt;
        for (int ky = 0; ky < 20; ++ky) {
            int vin  = v + ky;
            int base = ((tp * 2 + py) * 31 + vin) * 16384;
            const uint16_t* p0 = P1 + base + lo;
            const uint16_t* p1 = p0 + 128;          // +16 batches
            uint4 a0 = *(const uint4*)(p0);
            uint4 a1 = *(const uint4*)(p0 + 4096);
            uint4 a2 = *(const uint4*)(p0 + 8192);
            uint4 c0 = *(const uint4*)(p1);
            uint4 c1 = *(const uint4*)(p1 + 4096);
            uint4 c2 = *(const uint4*)(p1 + 8192);
            Frag16 af;
            const uint32_t* ap = Al + ((kt * 20 + ky) * 2 + py) * 256;
            af.q4[0] = *(const uint4*)(ap);
            af.q4[1] = *(const uint4*)(ap + 4);
            Frag16 b00, b01, b10, b11;
            b00.q4[0] = a0; b00.q4[1] = a1;         // u-tile 0 (s=0)
            b01.q4[0] = c0; b01.q4[1] = c1;
            b10.q4[0] = a1; b10.q4[1] = a2;         // u-tile 1 (s=8)
            b11.q4[0] = c1; b11.q4[1] = c2;
            acc[0][0] = WMMA16(af, b00, acc[0][0]);
            acc[0][1] = WMMA16(af, b01, acc[0][1]);
            acc[1][0] = WMMA16(af, b10, acc[1][0]);
            acc[1][1] = WMMA16(af, b11, acc[1][1]);
        }
    }

    // rows: oi = h (lane half), du = r  ->  o uniform per lane-half
    int o = 2 * h + (1 - py);
    int y = 2 * v + py;
    int hc_row = (o * 19 + (t + 7)) * 22 + (v + 5);
    int bn8 = (b0 + n) * 8;
#pragma unroll
    for (int ut = 0; ut < 2; ++ut) {
#pragma unroll
        for (int r = 0; r < 8; ++r) {
            if (ut == 1 && r >= 4) continue;        // u > 11 not stored
            int u = ut * 8 + r;
            int x = 2 * u;
            float mk = mask[((o * 10 + t) * 24 + y) * 24 + x];
            int wp0 = u + 5;
            int e0  = (hc_row * 3 + (wp0 >> 3)) * 4096 + bn8 + (wp0 & 7);
#pragma unroll
            for (int nb = 0; nb < 2; ++nb) {
                float val = fmaxf(acc[ut][nb][r], 0.0f) * mk;
                uint16_t hv = f32_to_f16u(val);
                H0[e0 + nb * 128] = hv;
                if (u >= 1) {
                    int wp1 = u - 1;
                    int e1 = (hc_row * 2 + (wp1 >> 3)) * 4096 + bn8 + (wp1 & 7);
                    H1[e1 + nb * 128] = hv;
                }
            }
        }
    }
}

// ---------------------------------------------------------------------------
// Conv2 implicit GEMM over compressed H. K segment = 2 kept-ky (lane halves
// read adjacent ykc rows) x 16 w-cols: per (c,kt,s): 4 B-b128 + 2 A-b128 ->
// 2 WMMAs. Scatter f32 to d_out (B,X,Y,T).
// grid = 480 row-tiles * 2 batch-groups; 8 waves * 32 batch.
// ---------------------------------------------------------------------------
__global__ void __launch_bounds__(256)
conv2_gemm(const uint16_t* __restrict__ H0, const uint16_t* __restrict__ H1,
           const uint32_t* __restrict__ A2, float* __restrict__ out) {
    int rt = blockIdx.x >> 1, bg = blockIdx.x & 1;
    int xt = rt % 2;
    int y  = (rt / 2) % 24;
    int t  = rt / 48;

    const uint16_t* Hc = xt ? H1 : H0;
    int wst = xt ? 2 : 3;                 // octets per (c,tp,ykc) row

    int l  = threadIdx.x & 31;
    int wv = threadIdx.x >> 5;
    int b0 = bg * 256 + wv * 32;
    int n  = l & 15, h = l >> 4;
    int bn8 = (b0 + n) * 8;

    v8f acc0 = {0.f,0.f,0.f,0.f,0.f,0.f,0.f,0.f};
    v8f acc1 = {0.f,0.f,0.f,0.f,0.f,0.f,0.f,0.f};
    const uint32_t* Al = A2 + l * 8;

    for (int c = 0; c < 4; ++c) {
        int pc = (c + 1) & 1;             // kept y parity for channel c
        int q  = (pc + y) & 1;            // kept ky parity for this output y
        int yb = (y + q - pc) >> 1;       // compressed y row base
        for (int kt = 0; kt < 10; ++kt) {
            int tp   = t + kt;
            int rowb = (c * 19 + tp) * 22;
            for (int s = 0; s < 5; ++s) {
                int ra = yb + 2 * s + h;  // lane-half selects ky pair member
                int e  = (rowb + ra) * wst * 4096 + bn8;
                const uint16_t* p = Hc + e;
                Frag16 bf0, bf1;
                bf0.q4[0] = *(const uint4*)(p);
                bf0.q4[1] = *(const uint4*)(p + 4096);
                bf1.q4[0] = *(const uint4*)(p + 128);
                bf1.q4[1] = *(const uint4*)(p + 128 + 4096);
                Frag16 af;
                const uint32_t* ap = Al + (((c * 10 + kt) * 5 + s) * 2 + q) * 256;
                af.q4[0] = *(const uint4*)(ap);
                af.q4[1] = *(const uint4*)(ap + 4);
                acc0 = WMMA16(af, bf0, acc0);
                acc1 = WMMA16(af, bf1, acc1);
            }
        }
    }

#pragma unroll
    for (int r = 0; r < 8; ++r) {
        int m = r + 8 * h;
        if (m < 12) {
            int x  = xt * 12 + m;
            int ob = (x * 24 + y) * 10 + t;
            out[(b0 + n) * 5760 + ob]      = acc0[r];
            out[(b0 + 16 + n) * 5760 + ob] = acc1[r];
        }
    }
}

// ---------------------------------------------------------------------------
extern "C" void kernel_launch(void* const* d_in, const int* in_sizes, int n_in,
                              void* d_out, int out_size, void* d_ws, size_t ws_size,
                              hipStream_t stream) {
    (void)in_sizes; (void)n_in; (void)out_size; (void)ws_size;
    const float* x    = (const float*)d_in[0];   // (512,24,24,10,1) f32
    const float* w1   = (const float*)d_in[1];   // (4,1,10,20,20)   f32
    const float* w2   = (const float*)d_in[2];   // (1,4,10,20,20)   f32
    const float* mask = (const float*)d_in[3];   // (4,10,24,24)     f32
    float* out = (float*)d_out;                  // (512,24,24,10,1) f32

    char* ws = (char*)d_ws;                      // needs ~108 MB
    uint16_t* P1 = (uint16_t*)(ws);
    uint16_t* H0 = (uint16_t*)(ws + H0_OFF);
    uint16_t* H1 = (uint16_t*)(ws + H1_OFF);
    uint32_t* A1 = (uint32_t*)(ws + A1_OFF);
    uint32_t* A2 = (uint32_t*)(ws + A2_OFF);

    // Zero all padded/compressed activation storage.
    hipMemsetAsync(ws, 0, MEMSET_BYTES, stream);

    prep_p1<<<5760, 256, 0, stream>>>(x, P1);
    prep_a1<<<400,  256, 0, stream>>>(w1, A1);
    prep_a2<<<400,  256, 0, stream>>>(w2, A2);
    conv1_gemm<<<480, 256, 0, stream>>>(P1, A1, mask, H0, H1);
    conv2_gemm<<<960, 256, 0, stream>>>(H0, H1, A2, out);
}
